// DGCNN_58677843198032
// MI455X (gfx1250) — matrix operand order repaired and checked by reading, not compiled, not run on hardware
//
#include <hip/hip_runtime.h>
#include <hip/hip_bf16.h>
#include <cstdint>
#include <cstddef>

// Problem constants (match reference)
#define NPG    256     // nodes per graph
#define EPG    4096    // edges per graph
#define NGRAPH 512
#define FIN    128
#define HDIM   64
#define TCLS   18
#define KTOP   16
#define DLAT   257
#define FSTR   264     // padded feat row stride (bf16 elems)
#define CPAD   288     // conv1 K padded to multiple of 32

typedef __attribute__((ext_vector_type(16))) __bf16 v16bf;
typedef __attribute__((ext_vector_type(8)))  float  v8f;

// ---------------- LDS layout (dynamic shared) ----------------
constexpr size_t A16(size_t v){ return (v + 15) & ~size_t(15); }
constexpr size_t OFF_X     = 0;                                    // bf16[256*128]
constexpr size_t OFF_FEAT  = A16(OFF_X     + (size_t)NPG*FIN*2);   // bf16[256*264]
constexpr size_t OFF_LIN   = A16(OFF_FEAT  + (size_t)NPG*FSTR*2);  // bf16[256*64] (reused: edge staging, topk)
constexpr size_t OFF_W     = A16(OFF_LIN   + (size_t)NPG*HDIM*2);  // bf16[64*128] (reused: conv1 w)
constexpr size_t OFF_INSRC = A16(OFF_W     + (size_t)HDIM*FIN*2);  // u16[4352]
constexpr size_t OFF_INOFF = A16(OFF_INSRC + (size_t)(EPG+NPG)*2); // i32[257]
constexpr size_t OFF_CNT   = A16(OFF_INOFF + (size_t)(NPG+1)*4);   // i32[256]
constexpr size_t OFF_OUTC  = A16(OFF_CNT   + (size_t)NPG*4);       // i32[256]
constexpr size_t OFF_INV   = A16(OFF_OUTC  + (size_t)NPG*4);       // f32[256]
constexpr size_t OFF_KEY   = A16(OFF_INV   + (size_t)NPG*4);       // f32[256]
constexpr size_t OFF_TMP   = A16(OFF_KEY   + (size_t)NPG*4);       // f32[256]
constexpr size_t OFF_C1    = A16(OFF_TMP   + (size_t)NPG*4);       // f32[256]
constexpr size_t OFF_P1    = A16(OFF_C1    + 256*4);               // f32[128]
constexpr size_t OFF_FLAT  = A16(OFF_P1    + 128*4);               // f32[128]
constexpr size_t OFF_LAST  = A16(OFF_FLAT  + 128*4);               // f32[32]
constexpr size_t OFF_OUTV  = A16(OFF_LAST  + 32*4);                // f32[32]
constexpr size_t LDS_BYTES = A16(OFF_OUTV  + 32*4);                // ~262 KB < 320 KB/WGP

// ---------------- WMMA fragment loaders (CDNA5 wave32 layouts) ----------------
// A: 16x32 bf16 tile, row-major LDS source. lane<16 -> K base 0, lane>=16 -> K base +8;
// elems 0..7 -> K[0..7], elems 8..15 -> K[16..23] (each +half*8).
__device__ __forceinline__ v16bf load_a_frag(const __bf16* base, int row0, int stride,
                                             int k0, int lane) {
  const int half = (lane >> 4) & 1;
  const int m    = lane & 15;
  const __bf16* rp = base + (size_t)(row0 + m) * stride + k0 + half * 8;
  v16bf a;
#pragma unroll
  for (int e = 0; e < 16; ++e)
    a[e] = rp[((e & 8) ? 16 : 0) + (e & 7)];
  return a;
}

// B: 32x16 bf16 tile with B[k][n] = W[n0+n][k0+k], W row-major (stride = row length).
// lane<16 -> K base 0, lane>=16 -> K base +8; elems 0..7 -> K[0..7], elems 8..15 -> K[16..23] (each +half*8).
__device__ __forceinline__ v16bf load_b_frag(const __bf16* w, int n0, int stride,
                                             int k0, int lane) {
  const int half = (lane >> 4) & 1;
  const int n    = lane & 15;
  const __bf16* rp = w + (size_t)(n0 + n) * stride + k0 + half * 8;
  v16bf b;
#pragma unroll
  for (int e = 0; e < 16; ++e) b[e] = rp[((e & 8) ? 16 : 0) + (e & 7)];
  return b;
}

// ---------------- One GCN layer: lin = in @ W^T + b (WMMA), then CSR message pass ----------------
template<int DIN, int DOUT>
__device__ __forceinline__ void gcn_layer(const __bf16* in, int instride,
                                          const float* __restrict__ Wg,
                                          const float* __restrict__ Bg,
                                          __bf16* sh_w, __bf16* sh_lin,
                                          __bf16* featout, int outstride,
                                          const unsigned short* insrc,
                                          const int* inoff,
                                          const float* invdeg,
                                          int tid, int wave, int lane) {
  // stage weights as bf16
  for (int i = tid; i < DOUT * DIN; i += 256) sh_w[i] = (__bf16)Wg[i];
  __syncthreads();

  const int half = (lane >> 4) & 1;
  const int nn   = lane & 15;
  constexpr int RT = NPG / 16, CT = DOUT / 16, TILES = RT * CT;
  for (int t = wave; t < TILES; t += 8) {        // wave-uniform: EXEC all ones for WMMA
    const int row0 = (t % RT) * 16;
    const int n0   = (t / RT) * 16;
    v8f c = {0.f, 0.f, 0.f, 0.f, 0.f, 0.f, 0.f, 0.f};
#pragma unroll
    for (int k0 = 0; k0 < DIN; k0 += 32) {
      v16bf a = load_a_frag(in, row0, instride, k0, lane);
      v16bf b = load_b_frag(sh_w, n0, DIN, k0, lane);
      c = __builtin_amdgcn_wmma_f32_16x16x32_bf16(false, a, false, b, (short)0, c,
                                                  false, false);
    }
    const float bias = Bg[n0 + nn];
#pragma unroll
    for (int r = 0; r < 8; ++r) {
      const int m = r + half * 8;  // D layout: VGPR r -> M=r (lanes<16) / M=8+r (lanes>=16)
      sh_lin[(size_t)(row0 + m) * DOUT + n0 + nn] = (__bf16)(c[r] + bias);
    }
  }
  __syncthreads();

  // message passing: one thread per destination node, deterministic serial reduce
  float acc[DOUT];
#pragma unroll
  for (int q = 0; q < DOUT; ++q) acc[q] = 0.f;
  const int beg = inoff[tid], end = inoff[tid + 1];
  for (int j = beg; j < end; ++j) {
    const unsigned s = insrc[j];
    const unsigned* row = (const unsigned*)(sh_lin + (size_t)s * DOUT);
#pragma unroll
    for (int q = 0; q < DOUT / 2; ++q) {
      const unsigned v = row[q];
      acc[2 * q]     += __builtin_bit_cast(float, v << 16);
      acc[2 * q + 1] += __builtin_bit_cast(float, v & 0xffff0000u);
    }
  }
  const float inv = invdeg[tid];
#pragma unroll
  for (int q = 0; q < DOUT; ++q)
    featout[(size_t)tid * outstride + q] = (__bf16)tanhf(acc[q] * inv);
  __syncthreads();
}

// ---------------- Fused per-graph DGCNN kernel ----------------
extern "C" __global__ void dgcnn_fused_graph_kernel_gfx1250(
    const float* __restrict__ x, const int* __restrict__ ei,
    const float* __restrict__ w0, const float* __restrict__ b0,
    const float* __restrict__ w1, const float* __restrict__ b1,
    const float* __restrict__ w2, const float* __restrict__ b2,
    const float* __restrict__ w3, const float* __restrict__ b3,
    const float* __restrict__ w4, const float* __restrict__ b4,
    const float* __restrict__ c1w, const float* __restrict__ c1b,
    const float* __restrict__ c2w, const float* __restrict__ c2b,
    const float* __restrict__ d1w, const float* __restrict__ d1b,
    const float* __restrict__ d2w, const float* __restrict__ d2b,
    float* __restrict__ out) {
  extern __shared__ char smem[];
  __bf16* sh_x    = (__bf16*)(smem + OFF_X);
  __bf16* sh_feat = (__bf16*)(smem + OFF_FEAT);
  __bf16* sh_lin  = (__bf16*)(smem + OFF_LIN);
  __bf16* sh_w    = (__bf16*)(smem + OFF_W);
  unsigned short* sh_insrc = (unsigned short*)(smem + OFF_INSRC);
  int*   sh_inoff = (int*)(smem + OFF_INOFF);
  int*   sh_cnt   = (int*)(smem + OFF_CNT);
  int*   sh_outc  = (int*)(smem + OFF_OUTC);
  float* sh_inv   = (float*)(smem + OFF_INV);
  float* sh_key   = (float*)(smem + OFF_KEY);
  float* sh_tmp   = (float*)(smem + OFF_TMP);
  float* sh_c1    = (float*)(smem + OFF_C1);
  float* sh_p1    = (float*)(smem + OFF_P1);
  float* sh_flat  = (float*)(smem + OFF_FLAT);
  float* sh_last  = (float*)(smem + OFF_LAST);
  float* sh_outv  = (float*)(smem + OFF_OUTV);

  const int g     = blockIdx.x;
  const int tid   = threadIdx.x;
  const int lane  = tid & 31;
  const int wave  = tid >> 5;
  const int nbase = g * NPG;
  const size_t E  = (size_t)NGRAPH * EPG;
  const int* srcp = ei + (size_t)g * EPG;
  const int* dstp = ei + E + (size_t)g * EPG;

  // -------- build CSR of incoming edges (+self loop), deterministic --------
  sh_cnt[tid]  = 0;
  sh_outc[tid] = 0;
  __syncthreads();

  int* sh_es = (int*)sh_lin;          // staged local src [4096]
  int* sh_ed = sh_es + EPG;           // staged local dst [4096]
  for (int e = tid; e < EPG; e += 256) {
    const int s = srcp[e] - nbase;
    const int d = dstp[e] - nbase;
    sh_es[e] = s;
    sh_ed[e] = d;
    atomicAdd(&sh_cnt[d], 1);
    atomicAdd(&sh_outc[s], 1);
  }
  // stage node features as bf16 (layer-0 A matrix)
  for (int i = tid; i < NPG * FIN; i += 256)
    sh_x[i] = (__bf16)x[(size_t)nbase * FIN + i];
  __syncthreads();

  if (tid == 0) {
    int run = 0;
    for (int n = 0; n < NPG; ++n) { sh_inoff[n] = run; run += sh_cnt[n] + 1; }
    sh_inoff[NPG] = run;
  }
  __syncthreads();

  {  // thread n collects its in-edges in edge order (deterministic), self-loop last
    int pos = sh_inoff[tid];
    for (int j = 0; j < EPG; ++j)
      if (sh_ed[j] == tid) sh_insrc[pos++] = (unsigned short)sh_es[j];
    sh_insrc[pos] = (unsigned short)tid;              // self loop
    sh_inv[tid]   = 1.0f / (float)(sh_outc[tid] + 1); // 1/deg[dst], deg counts self loop
  }
  __syncthreads();

  // -------- GCN layers 0..3 (WMMA bf16 -> f32) --------
  gcn_layer<FIN,  HDIM>(sh_x,               FIN,  w0, b0, sh_w, sh_lin,
                        sh_feat + 0 * HDIM, FSTR, sh_insrc, sh_inoff, sh_inv, tid, wave, lane);
  gcn_layer<HDIM, HDIM>(sh_feat + 0 * HDIM, FSTR, w1, b1, sh_w, sh_lin,
                        sh_feat + 1 * HDIM, FSTR, sh_insrc, sh_inoff, sh_inv, tid, wave, lane);
  gcn_layer<HDIM, HDIM>(sh_feat + 1 * HDIM, FSTR, w2, b2, sh_w, sh_lin,
                        sh_feat + 2 * HDIM, FSTR, sh_insrc, sh_inoff, sh_inv, tid, wave, lane);
  gcn_layer<HDIM, HDIM>(sh_feat + 2 * HDIM, FSTR, w3, b3, sh_w, sh_lin,
                        sh_feat + 3 * HDIM, FSTR, sh_insrc, sh_inoff, sh_inv, tid, wave, lane);

  // -------- layer 4 (64 -> 1), scalar --------
  {
    const __bf16* in4 = sh_feat + 3 * HDIM;
    float s = b4[0];
#pragma unroll
    for (int c = 0; c < HDIM; ++c)
      s += (float)in4[(size_t)tid * FSTR + c] * w4[c];
    sh_tmp[tid] = s;
    __syncthreads();
    float a = 0.f;
    for (int j = sh_inoff[tid]; j < sh_inoff[tid + 1]; ++j) a += sh_tmp[sh_insrc[j]];
    const float h5 = tanhf(a * sh_inv[tid]);
    sh_key[tid] = h5;
    sh_feat[(size_t)tid * FSTR + 256] = (__bf16)h5;
    __syncthreads();
  }

  // -------- global_sort_pool: rank by descending key, stable on index --------
  const float kn = sh_key[tid];
  int rank = 0;
  for (int j = 0; j < NPG; ++j) {
    const float kj = sh_key[j];
    rank += (kj > kn) || (kj == kn && j < tid);
  }
  __bf16* sh_topk = sh_lin;  // reuse: [16][288] bf16
  for (int i = tid; i < KTOP * CPAD; i += 256) sh_topk[i] = (__bf16)0.f;
  for (int i = tid; i < KTOP * CPAD; i += 256) {   // conv1 weights, K-padded
    const int o = i / CPAD, k = i % CPAD;
    sh_w[i] = (k < DLAT) ? (__bf16)c1w[o * DLAT + k] : (__bf16)0.f;
  }
  __syncthreads();
  if (rank < KTOP)
    for (int c = 0; c < DLAT; ++c)
      sh_topk[rank * CPAD + c] = sh_feat[(size_t)tid * FSTR + c];
  __syncthreads();

  // -------- conv1 == 16x16 tile, K=288 (9 WMMA steps), wave 0 only --------
  if (wave == 0) {
    v8f c = {0.f, 0.f, 0.f, 0.f, 0.f, 0.f, 0.f, 0.f};
#pragma unroll
    for (int k0 = 0; k0 < CPAD; k0 += 32) {
      v16bf a = load_a_frag(sh_topk, 0, CPAD, k0, lane);
      v16bf b = load_b_frag(sh_w,    0, CPAD, k0, lane);
      c = __builtin_amdgcn_wmma_f32_16x16x32_bf16(false, a, false, b, (short)0, c,
                                                  false, false);
    }
    const int half = (lane >> 4) & 1;
    const int n = lane & 15;            // out channel
    const float bias = c1b[n];
#pragma unroll
    for (int r = 0; r < 8; ++r) {
      const int m = r + half * 8;       // slot index
      sh_c1[n * 16 + m] = fmaxf(c[r] + bias, 0.f);   // relu, stored [och][slot]
    }
  }
  __syncthreads();

  // -------- maxpool1d(2,2): [16][16] -> [16][8] --------
  if (tid < 128) {
    const int o = tid >> 3, i2 = tid & 7;
    sh_p1[tid] = fmaxf(sh_c1[o * 16 + 2 * i2], sh_c1[o * 16 + 2 * i2 + 1]);
  }
  __syncthreads();

  // -------- conv2: 16->32 ch, kernel 5, len 8 -> 4 --------
  if (tid < 128) {
    const int oc = tid >> 2, t = tid & 3;
    float s = c2b[oc];
#pragma unroll
    for (int ic = 0; ic < 16; ++ic)
#pragma unroll
      for (int kk = 0; kk < 5; ++kk)
        s += sh_p1[ic * 8 + t + kk] * c2w[oc * 80 + ic * 5 + kk];
    sh_flat[oc * 4 + t] = fmaxf(s, 0.f);
  }
  __syncthreads();

  // -------- dense1 (128->32, relu) --------
  if (tid < 32) {
    float s = d1b[tid];
#pragma unroll
    for (int i = 0; i < 128; ++i) s += sh_flat[i] * d1w[tid * 128 + i];
    s = fmaxf(s, 0.f);
    sh_last[tid] = s;
    out[2 * NGRAPH * TCLS + g * 32 + tid] = s;   // third output: "last"
  }
  __syncthreads();

  // -------- dense2 (32->18) --------
  if (tid < TCLS) {
    float s = d2b[tid];
#pragma unroll
    for (int i = 0; i < 32; ++i) s += sh_last[i] * d2w[tid * 32 + i];
    sh_outv[tid] = s;
    out[NGRAPH * TCLS + g * TCLS + tid] = s;     // second output: raw logits
  }
  __syncthreads();

  // -------- log_softmax --------
  if (tid == 0) {
    float m = -1e30f;
    for (int t = 0; t < TCLS; ++t) m = fmaxf(m, sh_outv[t]);
    float s = 0.f;
    for (int t = 0; t < TCLS; ++t) s += expf(sh_outv[t] - m);
    const float lse = logf(s);
    for (int t = 0; t < TCLS; ++t)
      out[g * TCLS + t] = sh_outv[t] - m - lse;  // first output
  }
}

extern "C" void kernel_launch(void* const* d_in, const int* in_sizes, int n_in,
                              void* d_out, int out_size, void* d_ws, size_t ws_size,
                              hipStream_t stream) {
  (void)in_sizes; (void)n_in; (void)out_size; (void)d_ws; (void)ws_size;
  const float* x   = (const float*)d_in[0];
  const int*   ei  = (const int*)d_in[1];
  // d_in[2]=batch, d_in[3]=num_graphs: implied by fixed geometry
  const float* w0 = (const float*)d_in[4];  const float* b0 = (const float*)d_in[5];
  const float* w1 = (const float*)d_in[6];  const float* b1 = (const float*)d_in[7];
  const float* w2 = (const float*)d_in[8];  const float* b2 = (const float*)d_in[9];
  const float* w3 = (const float*)d_in[10]; const float* b3 = (const float*)d_in[11];
  const float* w4 = (const float*)d_in[12]; const float* b4 = (const float*)d_in[13];
  const float* c1w = (const float*)d_in[14]; const float* c1b = (const float*)d_in[15];
  const float* c2w = (const float*)d_in[16]; const float* c2b = (const float*)d_in[17];
  const float* d1w = (const float*)d_in[18]; const float* d1b = (const float*)d_in[19];
  const float* d2w = (const float*)d_in[20]; const float* d2b = (const float*)d_in[21];
  float* outp = (float*)d_out;

  hipFuncSetAttribute((const void*)dgcnn_fused_graph_kernel_gfx1250,
                      hipFuncAttributeMaxDynamicSharedMemorySize, (int)LDS_BYTES);
  dgcnn_fused_graph_kernel_gfx1250<<<NGRAPH, NPG, LDS_BYTES, stream>>>(
      x, ei, w0, b0, w1, b1, w2, b2, w3, b3, w4, b4,
      c1w, c1b, c2w, c2b, d1w, d1b, d2w, d2b, outp);
}
